// GraphAttention_46170898432679
// MI455X (gfx1250) — matrix operand order repaired and checked by reading, not compiled, not run on hardware
//
#include <hip/hip_runtime.h>
#include <hip/hip_bf16.h>

#define NNBR   64
#define HID    128
#define NHEAD  8
#define MROWS  80      // 65 rows padded to 5 tiles of 16 (per batch)
#define QPITCH 136     // halves per LDS row: 272B, conflict-free A-frag loads
#define NB     2       // batches per workgroup
#define MT     (NB * 5)  // M-tiles per GEMM (10)

typedef __attribute__((ext_vector_type(16))) _Float16 v16h;
typedef __attribute__((ext_vector_type(8)))  _Float16 h8;
typedef __attribute__((ext_vector_type(8)))  float    v8f;

union ABfrag { v16h v; h8 h[2]; };

// ---- prep: WaT[h][n][k] = (f16) Wa[h][k][n]  (B-matrix pre-transpose+convert) ----
__global__ __launch_bounds__(256) void ga_prep(const float* __restrict__ Wa,
                                               _Float16* __restrict__ WaT) {
    int i = blockIdx.x * 256 + threadIdx.x;          // 0 .. 8*128*128-1
    int h = i >> 14, rem = i & 16383, n = rem >> 7, k = rem & 127;
    WaT[i] = (_Float16)Wa[h * 16384 + k * 128 + n];
}

// ---- main: one workgroup per NB batch elements ----
__global__ __launch_bounds__(256) void ga_main(
        const float*    __restrict__ query,   // [B,64,128]
        const float*    __restrict__ value,   // [B,1,128]
        const _Float16* __restrict__ WaT,     // [8,128,128] (n,k) f16
        const float*    __restrict__ ba,      // [8,128]
        const float*    __restrict__ Wo,      // [1024,128]
        const float*    __restrict__ bo,      // [128]
        float*          __restrict__ out) {   // [B,1,128]
    __shared__ _Float16 qf[NB * MROWS * QPITCH];   // two 80x128 q matrices stacked
    __shared__ float vv[NB * HID];
    __shared__ float ctx[NB * NHEAD * HID];
    __shared__ float ew[NB * MROWS];
    __shared__ float scal[NB];

    const int b0   = blockIdx.x * NB;
    const int t    = threadIdx.x;
    const int wave = t >> 5;
    const int lane = t & 31;
    const int hl   = lane & 15;   // column / M-row within tile
    const int hi   = lane >> 4;   // half-wave select
    const int ncol = wave * 16 + hl;   // this wave owns output cols [16w,16w+16)
    const int p    = t >> 7;      // batch slot for epilogue phases
    const int tl   = t & 127;

    // ---- stage q (both batches) into LDS as f16 ----
    for (int i = t; i < NB * NNBR * HID; i += 256) {
        int pb = i >> 13, rem = i & 8191;
        int r = rem >> 7, c = rem & 127;
        qf[(pb * MROWS + r) * QPITCH + c] =
            (_Float16)query[(size_t)(b0 + pb) * (NNBR * HID) + rem];
    }
    for (int i = t; i < NB * HID; i += 256) {
        int pb = i >> 7, c = i & 127;
        float x = value[(size_t)(b0 + pb) * HID + c];
        qf[(pb * MROWS + 64) * QPITCH + c] = (_Float16)x;
        vv[i] = x;                                   // original value vectors (f32)
    }
    for (int i = t; i < NB * 15 * QPITCH; i += 256) { // zero-pad rows 65..79 each batch
        int pb = i / (15 * QPITCH), rem = i % (15 * QPITCH);
        qf[(pb * MROWS + 65) * QPITCH + rem] = (_Float16)0.f;
    }
    __syncthreads();

    for (int head = 0; head < NHEAD; ++head) {
        // ---- q_new(160x128) = q(160x128) @ Wa[head](128x128), f16 in / f32 accum ----
        const _Float16* wt = WaT + head * (HID * HID) + ncol * HID;
        ABfrag bf[4];
        #pragma unroll
        for (int ks = 0; ks < 4; ++ks) {             // B: col=ncol, halves0-7=K+8hi+0..7, 8-15=K+8hi+16..23
            const h8* pw = (const h8*)(wt + ks * 32 + hi * 8);
            bf[ks].h[0] = pw[0];
            bf[ks].h[1] = pw[2];
        }
        v8f acc[MT] = {};
        #pragma unroll
        for (int m = 0; m < MT; ++m) {
            const _Float16* arow = qf + (m * 16 + hl) * QPITCH + hi * 8;
            #pragma unroll
            for (int ks = 0; ks < 4; ++ks) {         // A: halves0-7=K+0..7, 8-15=K+16..23
                const h8* pa = (const h8*)(arow + ks * 32);
                ABfrag af;
                af.h[0] = pa[0];
                af.h[1] = pa[2];                     // +16 halves
                acc[m] = __builtin_amdgcn_wmma_f32_16x16x32_f16(
                    false, af.v, false, bf[ks].v, (short)0, acc[m], false, false);
            }
        }
        __syncthreads();                             // all A reads done before overwrite

        float bias = ba[head * HID + ncol];
        #pragma unroll
        for (int m = 0; m < MT; ++m)
            #pragma unroll
            for (int v = 0; v < 8; ++v) {            // C: row = 16m + v + 8*hi, col = ncol
                int row = m * 16 + v + hi * 8;       // 0..159 spans both batches
                qf[row * QPITCH + ncol] = (_Float16)(acc[m][v] + bias);
            }
        __syncthreads();

        // ---- e[j] = value . q_new[j]  (both batches in parallel) ----
        if (tl < NNBR + 1) {
            float e = 0.f;
            for (int c = 0; c < HID; ++c)
                e += vv[p * HID + c] * (float)qf[(p * MROWS + tl) * QPITCH + c];
            ew[p * MROWS + tl] = e;
        }
        __syncthreads();
        // ---- softmax (65 values, serial per batch slot) ----
        if (tl == 0) {
            float mx = ew[p * MROWS];
            for (int j = 1; j < NNBR + 1; ++j) mx = fmaxf(mx, ew[p * MROWS + j]);
            float s = 0.f;
            for (int j = 0; j < NNBR + 1; ++j) {
                float ex = __expf(ew[p * MROWS + j] - mx);
                ew[p * MROWS + j] = ex;
                s += ex;
            }
            scal[p] = 1.f / s;
        }
        __syncthreads();
        // ---- context[c] = (1/sum) * sum_j exp[j] * q_new[j][c] ----
        {
            float ssum = 0.f;
            for (int j = 0; j < NNBR + 1; ++j)
                ssum += ew[p * MROWS + j] * (float)qf[(p * MROWS + j) * QPITCH + tl];
            ctx[(p * NHEAD + head) * HID + tl] = ssum * scal[p];
        }
        __syncthreads();
    }

    // ---- out = relu(ctx @ Wo + bo), one thread per (batch, channel) ----
    {
        float a = 0.f;
        for (int c = 0; c < NHEAD * HID; ++c)
            a += ctx[p * (NHEAD * HID) + c] * Wo[c * HID + tl];
        a += bo[tl];
        out[(size_t)(b0 + p) * HID + tl] = fmaxf(a, 0.f);
    }
}

extern "C" void kernel_launch(void* const* d_in, const int* in_sizes, int n_in,
                              void* d_out, int out_size, void* d_ws, size_t ws_size,
                              hipStream_t stream) {
    const float* query = (const float*)d_in[0];   // [8192,64,128]
    const float* value = (const float*)d_in[1];   // [8192,1,128]
    const float* Wa    = (const float*)d_in[2];   // [8,128,128]
    const float* ba    = (const float*)d_in[3];   // [8,128]
    const float* Wo    = (const float*)d_in[4];   // [1024,128]
    const float* bo    = (const float*)d_in[5];   // [128]
    float* out = (float*)d_out;

    _Float16* WaT = (_Float16*)d_ws;              // 8*128*128*2 = 256 KB
    (void)in_sizes; (void)n_in; (void)out_size; (void)ws_size;

    ga_prep<<<(NHEAD * HID * HID) / 256, 256, 0, stream>>>(Wa, WaT);
    ga_main<<<8192 / NB, 256, 0, stream>>>(query, value, WaT, ba, Wo, bo, out);
}
